// KalmanVAE_1297080123809
// MI455X (gfx1250) — matrix operand order repaired and checked by reading, not compiled, not run on hardware
//
#include <hip/hip_runtime.h>
#include <hip/hip_bf16.h>
#include <math.h>
#include <stdint.h>

typedef __bf16 bf16_t;
typedef __attribute__((ext_vector_type(16))) __bf16 v16bf;
typedef __attribute__((ext_vector_type(8)))  float  v8f;

#define BSZ  32
#define TSQ  50
#define BT   1600
#define OBS  32
#define LAT  16
#define MMIX 16

#define LDT 40   // LDS tile stride in bf16 (80B rows: 16B aligned, conflict-free)

__device__ inline v8f zero8() {
  v8f z;
#pragma unroll
  for (int i = 0; i < 8; ++i) z[i] = 0.0f;
  return z;
}

// ---- gfx1250 async global->LDS copy (ASYNCcnt) ----
__device__ inline void async_ld_b128(unsigned lds_off, const void* gaddr) {
  asm volatile("global_load_async_to_lds_b128 %0, %1, off"
               :: "v"(lds_off), "v"((unsigned long long)(size_t)gaddr)
               : "memory");
}
__device__ inline void wait_async0() {
  asm volatile("s_wait_asynccnt 0x0" ::: "memory");
}
__device__ inline unsigned lds_off_of(const void* p) {
  return (unsigned)(size_t)p;   // low 32 bits of LDS aperture address = LDS offset
}

// ---- WMMA fragment loaders (contiguous per CDNA5 16-bit A/B VGPR layouts) ----
// A (16x32): lane(hi,r) needs K = hi*8 + {0..7} and 16 + hi*8 + {0..7}  (2x b128)
__device__ inline v16bf fragA(const bf16_t* At, int mBase, int hi, int r) {
  const bf16_t* rp = At + (mBase + r) * LDT + hi * 8;
  v16bf a;
#pragma unroll
  for (int e = 0; e < 8; ++e) { a[e] = rp[e]; a[e + 8] = rp[16 + e]; }
  return a;
}
// B (32x16), tile stored [n][k]: lane(hi,c) needs K = hi*8 + {0..7} and 16 + hi*8 + {0..7} (2x b128)
__device__ inline v16bf fragB(const bf16_t* Bt, int nBase, int hi, int c) {
  const bf16_t* cp = Bt + (nBase + c) * LDT + hi * 8;
  v16bf b;
#pragma unroll
  for (int e = 0; e < 8; ++e) { b[e] = cp[e]; b[e + 8] = cp[16 + e]; }
  return b;
}

// ---------------------------------------------------------------------
// Implicit-GEMM with bf16 WMMA.  B weights are [N][K] (row-major by N).
// AMODE: 0 dense A[M,K]; 1 conv gather (k4,s2,pad1); 2 deconv gather
// ACT: 0 none, 1 relu, 2 sigmoid   STOREF: 0 bf16 NCHW, 1 f32
// 128 threads (4 waves), block tile 64x64, wave tile 32x32.
// ---------------------------------------------------------------------
template <int AMODE, int ACT, int STOREF>
__global__ void __launch_bounds__(128)
gemm_wmma(const bf16_t* __restrict__ A, const bf16_t* __restrict__ Bw,
          const float* __restrict__ bias, bf16_t* __restrict__ outb,
          float* __restrict__ outf, int Mdim, int Ndim, int Kdim,
          int Cin, int IH, int IW, int OH, int OW) {
  __shared__ bf16_t Atile[64 * LDT];
  __shared__ bf16_t Btile[64 * LDT];
  const int tid = threadIdx.x;
  const int lane = tid & 31;
  const int wv = tid >> 5;
  const int hi = lane >> 4;
  const int ln16 = lane & 15;
  const int blockM = blockIdx.y * 64;
  const int blockN = blockIdx.x * 64;
  const int mW = (wv >> 1) * 32;
  const int nW = (wv & 1) * 32;
  const int OHW = OH * OW;
  const bool bFull = (blockN + 64 <= Ndim);

  v8f c00 = zero8(), c01 = zero8(), c10 = zero8(), c11 = zero8();

  for (int k0 = 0; k0 < Kdim; k0 += 32) {
    __syncthreads();
    const bool kFull = (k0 + 32 <= Kdim);
    // ================= A tile =================
    if (AMODE == 0 && kFull && (blockM + 64 <= Mdim)) {
      // async DMA: 64 rows x 32 bf16, 16B segments, 2 per thread
#pragma unroll
      for (int it = 0; it < 2; ++it) {
        int e = tid + it * 128;           // 0..255
        int row = e >> 2, seg = e & 3;
        const bf16_t* gp = A + (size_t)(blockM + row) * Kdim + k0 + seg * 8;
        async_ld_b128(lds_off_of(Atile + row * LDT + seg * 8), gp);
      }
    } else {
#pragma unroll
      for (int it = 0; it < 16; ++it) {
        int e = tid * 16 + it;            // 64x32 tile
        int row = e >> 5, kk = e & 31;
        int m = blockM + row;
        int k = k0 + kk;
        bf16_t val;
        bool ok = (m < Mdim) && (k < Kdim);
        int mm = (m < Mdim) ? m : (Mdim - 1);
        int kc = (k < Kdim) ? k : 0;
        if (AMODE == 0) {
          val = A[(size_t)mm * Kdim + kc];
        } else {
          int img = mm / OHW;
          int ryx = mm - img * OHW;
          int oy = ryx / OW, ox = ryx - oy * OW;
          int ci = kc >> 4;
          int ky = (kc >> 2) & 3, kx = kc & 3;
          int iy, ix;
          if (AMODE == 1) {               // forward conv s2 k4 pad1
            iy = oy * 2 + ky - 1;
            ix = ox * 2 + kx - 1;
            ok = ok && ((unsigned)iy < (unsigned)IH) && ((unsigned)ix < (unsigned)IW);
          } else {                        // conv_transpose s2 k4 SAME
            int ty = oy + ky - 2, tx = ox + kx - 2;
            iy = ty >> 1; ix = tx >> 1;
            ok = ok && (ty >= 0) && (tx >= 0) && !(ty & 1) && !(tx & 1) &&
                 (iy < IH) && (ix < IW);
          }
          int iyc = ok ? iy : 0, ixc = ok ? ix : 0;
          val = A[(((size_t)img * Cin + ci) * IH + iyc) * IW + ixc];
        }
        Atile[row * LDT + kk] = ok ? val : (bf16_t)0.0f;
      }
    }
    // ================= B tile ([n][k]) =================
    if (bFull && kFull) {
#pragma unroll
      for (int it = 0; it < 2; ++it) {
        int e = tid + it * 128;
        int row = e >> 2, seg = e & 3;
        const bf16_t* gp = Bw + (size_t)(blockN + row) * Kdim + k0 + seg * 8;
        async_ld_b128(lds_off_of(Btile + row * LDT + seg * 8), gp);
      }
    } else {
#pragma unroll
      for (int it = 0; it < 16; ++it) {
        int e = tid * 16 + it;
        int nn = e >> 5, kk = e & 31;
        int n = blockN + nn, k = k0 + kk;
        bf16_t val = (bf16_t)0.0f;
        if (n < Ndim && k < Kdim) val = Bw[(size_t)n * Kdim + k];
        Btile[nn * LDT + kk] = val;
      }
    }
    if (tid == 0 && (k0 + 32) < Kdim)
      __builtin_prefetch(&Bw[(size_t)blockN * Kdim + k0 + 32], 0, 1);
    wait_async0();
    __syncthreads();

    v16bf a0 = fragA(Atile, mW, hi, ln16);
    v16bf a1 = fragA(Atile, mW + 16, hi, ln16);
    v16bf b0 = fragB(Btile, nW, hi, ln16);
    v16bf b1 = fragB(Btile, nW + 16, hi, ln16);
    c00 = __builtin_amdgcn_wmma_f32_16x16x32_bf16(false, a0, false, b0, (short)0, c00, false, false);
    c01 = __builtin_amdgcn_wmma_f32_16x16x32_bf16(false, a0, false, b1, (short)0, c01, false, false);
    c10 = __builtin_amdgcn_wmma_f32_16x16x32_bf16(false, a1, false, b0, (short)0, c10, false, false);
    c11 = __builtin_amdgcn_wmma_f32_16x16x32_bf16(false, a1, false, b1, (short)0, c11, false, false);
  }

  // ---- epilogue: bias + activation + scatter ----
  auto emit = [&](const v8f& cc, int mS, int nS) {
#pragma unroll
    for (int r = 0; r < 8; ++r) {
      int m = blockM + mS + r + (hi ? 8 : 0);
      int n = blockN + nS + ln16;
      if (m < Mdim && n < Ndim) {
        float v = cc[r] + bias[n];
        if (ACT == 1) v = fmaxf(v, 0.0f);
        if (ACT == 2) v = 1.0f / (1.0f + __expf(-v));
        int img = m / OHW;
        int ryx = m - img * OHW;
        int oy = ryx / OW, ox = ryx - oy * OW;
        size_t idx = (((size_t)img * Ndim + n) * OH + oy) * OW + ox;
        if (STOREF) outf[idx] = v;
        else        outb[idx] = (bf16_t)v;
      }
    }
  };
  emit(c00, mW, nW);
  emit(c01, mW, nW + 16);
  emit(c10, mW + 16, nW);
  emit(c11, mW + 16, nW + 16);
}

// ------------------------- prep kernels -------------------------
__global__ void cvt_f32_bf16(const float* __restrict__ src, bf16_t* __restrict__ dst, int n) {
  int i = blockIdx.x * blockDim.x + threadIdx.x;
  if (i < n) dst[i] = (bf16_t)src[i];
}

// deconv weight (O,I,4,4) -> bf16 [O][I*16] with spatial flip
__global__ void prep_flipw(const float* __restrict__ w, bf16_t* __restrict__ wT,
                           int Cin, int Cout) {
  int i = blockIdx.x * blockDim.x + threadIdx.x;
  int total = Cout * Cin * 16;
  if (i >= total) return;
  int co = i / (Cin * 16);
  int rem = i - co * Cin * 16;
  int ci = rem >> 4;
  int ky = (rem >> 2) & 3, kx = rem & 3;
  float v = w[(((size_t)co * Cin + ci) * 4 + (3 - ky)) * 4 + (3 - kx)];
  wT[(size_t)co * (Cin * 16) + ci * 16 + ky * 4 + kx] = (bf16_t)v;
}

__global__ void sample_kernel(const float* __restrict__ amu, const float* __restrict__ alv,
                              const float* __restrict__ eps, float* __restrict__ a_sample,
                              bf16_t* __restrict__ a_bf, int n) {
  int i = blockIdx.x * blockDim.x + threadIdx.x;
  if (i >= n) return;
  float sd = fminf(__expf(0.5f * alv[i]), 100.0f);
  float s = amu[i] + sd * eps[i];
  a_sample[i] = s;
  a_bf[i] = (bf16_t)s;
}

// ------------------------- LSTM over joint sequence -------------------------
__global__ void __launch_bounds__(256)
lstm_kernel(const float* __restrict__ a_sample, const float* __restrict__ Wih,
            const float* __restrict__ Whh, const float* __restrict__ bih,
            const float* __restrict__ bhh, const float* __restrict__ start_code,
            float* __restrict__ dyn) {
  __shared__ float hh[BSZ * MMIX], cc[BSZ * MMIX], xt[BSZ * OBS], g[BSZ * 64];
  int tid = threadIdx.x;
  for (int i = tid; i < BSZ * MMIX; i += 256) { hh[i] = 0.f; cc[i] = 0.f; }
  __syncthreads();
  for (int s = 0; s <= TSQ; ++s) {
    for (int i = tid; i < BSZ * OBS; i += 256) {
      int b = i >> 5, k = i & 31;
      xt[i] = (s == 0) ? start_code[k] : a_sample[((size_t)b * TSQ + (s - 1)) * OBS + k];
    }
    __syncthreads();
    for (int i = tid; i < BSZ * 64; i += 256) {
      int b = i >> 6, j = i & 63;
      float acc = bih[j] + bhh[j];
      for (int k = 0; k < OBS; ++k) acc += xt[b * OBS + k] * Wih[j * OBS + k];
      for (int k = 0; k < MMIX; ++k) acc += hh[b * MMIX + k] * Whh[j * MMIX + k];
      g[i] = acc;
    }
    __syncthreads();
    for (int i = tid; i < BSZ * MMIX; i += 256) {
      int b = i >> 4, m = i & 15;
      float gi = g[b * 64 + m], gf = g[b * 64 + 16 + m];
      float gg = g[b * 64 + 32 + m], go = g[b * 64 + 48 + m];
      float si = 1.f / (1.f + __expf(-gi));
      float sf = 1.f / (1.f + __expf(-gf));
      float so = 1.f / (1.f + __expf(-go));
      float c2 = sf * cc[i] + si * tanhf(gg);
      float h2 = so * tanhf(c2);
      cc[i] = c2; hh[i] = h2;
      dyn[(size_t)s * BSZ * MMIX + i] = h2;
    }
    __syncthreads();
  }
}

// softmax over M=16 + mixture of A (16x16x16) and C (16x32x16)
__global__ void __launch_bounds__(256)
mix_kernel(const float* __restrict__ dyn, const float* __restrict__ Am,
           const float* __restrict__ Cm, float* __restrict__ A_t, float* __restrict__ C_t) {
  int tb = blockIdx.x;      // t*32 + b, t < 50
  int tid = threadIdx.x;
  __shared__ float w[MMIX];
  if (tid < MMIX) w[tid] = dyn[(size_t)tb * MMIX + tid];
  __syncthreads();
  if (tid == 0) {
    float mx = w[0];
    for (int k = 1; k < MMIX; ++k) mx = fmaxf(mx, w[k]);
    float sm = 0.f;
    for (int k = 0; k < MMIX; ++k) { w[k] = __expf(w[k] - mx); sm += w[k]; }
    float inv = 1.f / sm;
    for (int k = 0; k < MMIX; ++k) w[k] *= inv;
  }
  __syncthreads();
  {
    float acc = 0.f;
    for (int m = 0; m < MMIX; ++m) acc += w[m] * Am[m * 256 + tid];
    A_t[(size_t)tb * 256 + tid] = acc;
  }
  for (int i = tid; i < 512; i += 256) {
    float acc = 0.f;
    for (int m = 0; m < MMIX; ++m) acc += w[m] * Cm[m * 512 + i];
    C_t[(size_t)tb * 512 + i] = acc;
  }
}

// ------------------------- Kalman filter (one block per batch) -------------------------
__global__ void __launch_bounds__(256)
kalman_kernel(const float* __restrict__ a_sample, const float* __restrict__ A_t,
              const float* __restrict__ C_t, const float* __restrict__ Qm,
              const float* __restrict__ Rm, const float* __restrict__ mu1,
              const float* __restrict__ Sig1, float* __restrict__ mu_f,
              float* __restrict__ Sig_f, float* __restrict__ mu_p,
              float* __restrict__ Sig_p) {
  int b = blockIdx.x, tid = threadIdx.x;
  __shared__ float mu[16], Sig[256], Ct[512], tmp[512], SA[2048], fac[32];
  __shared__ float Km[512], muz[16], KC[256], Sigz[256], Atl[256], ASig[256], rv[32];
  if (tid < 16) mu[tid] = mu1[tid];
  Sig[tid] = Sig1[tid];
  __syncthreads();
  for (int t = 0; t < TSQ; ++t) {
    int tb = t * BSZ + b;
    for (int i = tid; i < 512; i += 256) Ct[i] = C_t[(size_t)tb * 512 + i];
    if (tid < 16) mu_p[(size_t)tb * 16 + tid] = mu[tid];
    Sig_p[(size_t)tb * 256 + tid] = Sig[tid];
    __syncthreads();
    if (tid < 32) {
      float acc = a_sample[((size_t)b * TSQ + t) * OBS + tid];
      for (int j = 0; j < 16; ++j) acc -= Ct[tid * 16 + j] * mu[j];
      rv[tid] = acc;
    }
    for (int i = tid; i < 512; i += 256) {       // tmp = Ct @ Sig (32x16)
      int r_ = i >> 4, c_ = i & 15;
      float acc = 0.f;
      for (int k = 0; k < 16; ++k) acc += Ct[r_ * 16 + k] * Sig[k * 16 + c_];
      tmp[i] = acc;
    }
    __syncthreads();
    for (int i = tid; i < 1024; i += 256) {      // SA = [S | I], S = tmp@Ct^T + R
      int r_ = i >> 5, c_ = i & 31;
      float acc = Rm[r_ * 32 + c_];
      for (int k = 0; k < 16; ++k) acc += tmp[r_ * 16 + k] * Ct[c_ * 16 + k];
      SA[r_ * 64 + c_] = acc;
      SA[r_ * 64 + 32 + c_] = (r_ == c_) ? 1.f : 0.f;
    }
    __syncthreads();
    for (int p = 0; p < 32; ++p) {               // Gauss-Jordan inverse of S
      if (tid < 32) fac[tid] = SA[tid * 64 + p];
      __syncthreads();
      float pinv = 1.f / fac[p];
      if (tid < 64) SA[p * 64 + tid] *= pinv;
      __syncthreads();
      for (int i = tid; i < 2048; i += 256) {
        int r_ = i >> 6, c_ = i & 63;
        if (r_ != p) SA[i] -= fac[r_] * SA[p * 64 + c_];
      }
      __syncthreads();
    }
    for (int i = tid; i < 512; i += 256) {       // K = Sig@Ct^T@Sinv = tmp^T @ Sinv
      int r_ = i >> 5, c_ = i & 31;
      float acc = 0.f;
      for (int k = 0; k < 32; ++k) acc += tmp[k * 16 + r_] * SA[k * 64 + 32 + c_];
      Km[i] = acc;
    }
    __syncthreads();
    if (tid < 16) {                              // mu_z = mu + K r
      float acc = mu[tid];
      for (int k = 0; k < 32; ++k) acc += Km[tid * 32 + k] * rv[k];
      muz[tid] = acc;
      mu_f[(size_t)tb * 16 + tid] = acc;
    }
    {                                            // KC = K @ Ct
      int r_ = tid >> 4, c_ = tid & 15;
      float acc = 0.f;
      for (int k = 0; k < 32; ++k) acc += Km[r_ * 32 + k] * Ct[k * 16 + c_];
      KC[tid] = acc;
    }
    __syncthreads();
    {                                            // Sig_z = Sig - KC@Sig
      int r_ = tid >> 4, c_ = tid & 15;
      float acc = Sig[tid];
      for (int k = 0; k < 16; ++k) acc -= KC[r_ * 16 + k] * Sig[k * 16 + c_];
      Sigz[tid] = acc;
      Sig_f[(size_t)tb * 256 + tid] = acc;
    }
    Atl[tid] = A_t[(size_t)tb * 256 + tid];
    __syncthreads();
    if (tid < 16) {                              // mu = At @ mu_z
      float acc = 0.f;
      for (int k = 0; k < 16; ++k) acc += Atl[tid * 16 + k] * muz[k];
      mu[tid] = acc;
    }
    {                                            // ASig = At @ Sig_z
      int r_ = tid >> 4, c_ = tid & 15;
      float acc = 0.f;
      for (int k = 0; k < 16; ++k) acc += Atl[r_ * 16 + k] * Sigz[k * 16 + c_];
      ASig[tid] = acc;
    }
    __syncthreads();
    {                                            // Sig = ASig@At^T + Q
      int r_ = tid >> 4, c_ = tid & 15;
      float acc = Qm[tid];
      for (int k = 0; k < 16; ++k) acc += ASig[r_ * 16 + k] * Atl[c_ * 16 + k];
      Sig[tid] = acc;
    }
    __syncthreads();
  }
}

// ------------------------- RTS smoother (reverse scan) -------------------------
__global__ void __launch_bounds__(256)
smooth_kernel(const float* __restrict__ mu_f, const float* __restrict__ Sig_f,
              const float* __restrict__ mu_p, const float* __restrict__ Sig_p,
              const float* __restrict__ A_t, float* __restrict__ mu_sm,
              float* __restrict__ Sig_sm) {
  int b = blockIdx.x, tid = threadIdx.x;
  __shared__ float mus[16], Sigs[256], GA[16 * 32], fac[16];
  __shared__ float tmpA[256], Jm[256], JD[256], Dm[256], Atl[256], Sfl[256], dmu[16];
  int tbL = (TSQ - 1) * BSZ + b;
  if (tid < 16) { mus[tid] = mu_f[(size_t)tbL * 16 + tid]; mu_sm[(size_t)tbL * 16 + tid] = mus[tid]; }
  Sigs[tid] = Sig_f[(size_t)tbL * 256 + tid];
  Sig_sm[(size_t)tbL * 256 + tid] = Sigs[tid];
  __syncthreads();
  for (int t = TSQ - 2; t >= 0; --t) {
    int tb0 = t * BSZ + b, tb1 = (t + 1) * BSZ + b;
    {
      int r_ = tid >> 4, c_ = tid & 15;
      GA[r_ * 32 + c_] = Sig_p[(size_t)tb0 * 256 + tid];
      GA[r_ * 32 + 16 + c_] = (r_ == c_) ? 1.f : 0.f;
    }
    Atl[tid] = A_t[(size_t)tb0 * 256 + tid];
    Sfl[tid] = Sig_f[(size_t)tb0 * 256 + tid];
    __syncthreads();
    for (int p = 0; p < 16; ++p) {               // invert Sig_p[t]
      if (tid < 16) fac[tid] = GA[tid * 32 + p];
      __syncthreads();
      float pinv = 1.f / fac[p];
      if (tid < 32) GA[p * 32 + tid] *= pinv;
      __syncthreads();
      for (int i = tid; i < 512; i += 256) {
        int r_ = i >> 5, c_ = i & 31;
        if (r_ != p) GA[i] -= fac[r_] * GA[p * 32 + c_];
      }
      __syncthreads();
    }
    {                                            // tmpA = Sig_f[t] @ At^T
      int r_ = tid >> 4, c_ = tid & 15;
      float acc = 0.f;
      for (int k = 0; k < 16; ++k) acc += Sfl[r_ * 16 + k] * Atl[c_ * 16 + k];
      tmpA[tid] = acc;
    }
    __syncthreads();
    {                                            // J = tmpA @ inv(Sig_p[t])
      int r_ = tid >> 4, c_ = tid & 15;
      float acc = 0.f;
      for (int k = 0; k < 16; ++k) acc += tmpA[r_ * 16 + k] * GA[k * 32 + 16 + c_];
      Jm[tid] = acc;
    }
    Dm[tid] = Sigs[tid] - Sig_p[(size_t)tb1 * 256 + tid];
    if (tid < 16) dmu[tid] = mus[tid] - mu_p[(size_t)tb1 * 16 + tid];
    __syncthreads();
    {                                            // JD = J @ D
      int r_ = tid >> 4, c_ = tid & 15;
      float acc = 0.f;
      for (int k = 0; k < 16; ++k) acc += Jm[r_ * 16 + k] * Dm[k * 16 + c_];
      JD[tid] = acc;
    }
    __syncthreads();
    {                                            // Sig_s = Sig_f + JD@J^T
      int r_ = tid >> 4, c_ = tid & 15;
      float acc = Sfl[tid];
      for (int k = 0; k < 16; ++k) acc += JD[r_ * 16 + k] * Jm[c_ * 16 + k];
      Sigs[tid] = acc;
      Sig_sm[(size_t)tb0 * 256 + tid] = acc;
    }
    if (tid < 16) {                              // mu_s = mu_f + J@dmu
      float acc = mu_f[(size_t)tb0 * 16 + tid];
      for (int k = 0; k < 16; ++k) acc += Jm[tid * 16 + k] * dmu[k];
      mus[tid] = acc;
      mu_sm[(size_t)tb0 * 16 + tid] = acc;
    }
    __syncthreads();
  }
}

static inline int cdiv(int a, int b) { return (a + b - 1) / b; }

extern "C" void kernel_launch(void* const* d_in, const int* in_sizes, int n_in,
                              void* d_out, int out_size, void* d_ws, size_t ws_size,
                              hipStream_t stream) {
  (void)in_sizes; (void)n_in; (void)out_size; (void)ws_size;
  const float* x    = (const float*)d_in[0];
  const float* eps  = (const float*)d_in[1];
  const float* ew[4] = {(const float*)d_in[2], (const float*)d_in[4], (const float*)d_in[6], (const float*)d_in[8]};
  const float* eb[4] = {(const float*)d_in[3], (const float*)d_in[5], (const float*)d_in[7], (const float*)d_in[9]};
  const float* wmu = (const float*)d_in[10]; const float* bmu = (const float*)d_in[11];
  const float* wlv = (const float*)d_in[12]; const float* blv = (const float*)d_in[13];
  const float* wfc = (const float*)d_in[14]; const float* bfc = (const float*)d_in[15];
  const float* dw[4] = {(const float*)d_in[16], (const float*)d_in[18], (const float*)d_in[20], (const float*)d_in[22]};
  const float* db[4] = {(const float*)d_in[17], (const float*)d_in[19], (const float*)d_in[21], (const float*)d_in[23]};
  const float* Wih = (const float*)d_in[24]; const float* Whh = (const float*)d_in[25];
  const float* bih = (const float*)d_in[26]; const float* bhh = (const float*)d_in[27];
  const float* start = (const float*)d_in[28];
  const float* mu1 = (const float*)d_in[29]; const float* Sg1 = (const float*)d_in[30];
  const float* Amat = (const float*)d_in[31]; const float* Cmat = (const float*)d_in[32];
  const float* Qm = (const float*)d_in[33]; const float* Rm = (const float*)d_in[34];

  float* out  = (float*)d_out;
  float* xhat = out;                 // 1,638,400
  float* amu  = out + 1638400;       // 51,200
  float* alv  = amu + 51200;         // 51,200
  float* musm = alv + 51200;         // 25,600
  float* sgsm = musm + 25600;        // 409,600

  char* wsp = (char*)d_ws;
  auto alloc = [&](size_t bytes) -> void* {
    void* p = (void*)wsp;
    wsp += (bytes + 255) & ~(size_t)255;
    return p;
  };
  bf16_t* xb  = (bf16_t*)alloc((size_t)1638400 * 2);
  bf16_t* e1  = (bf16_t*)alloc((size_t)1600 * 32 * 256 * 2);
  bf16_t* e2  = (bf16_t*)alloc((size_t)1600 * 64 * 64 * 2);
  bf16_t* e3  = (bf16_t*)alloc((size_t)1600 * 128 * 16 * 2);
  bf16_t* e4  = (bf16_t*)alloc((size_t)1600 * 256 * 4 * 2);
  bf16_t* abf = (bf16_t*)alloc((size_t)51200 * 2);
  bf16_t* d0  = (bf16_t*)alloc((size_t)1600 * 1024 * 2);
  bf16_t* d1  = (bf16_t*)alloc((size_t)1600 * 128 * 16 * 2);
  bf16_t* d2  = (bf16_t*)alloc((size_t)1600 * 64 * 64 * 2);
  bf16_t* d3  = (bf16_t*)alloc((size_t)1600 * 32 * 256 * 2);
  // weights, all [N][K] bf16 (K = Cin*16 for convs, contiguous)
  bf16_t* wc0 = (bf16_t*)alloc((size_t)32 * 16 * 2);
  bf16_t* wc1 = (bf16_t*)alloc((size_t)64 * 512 * 2);
  bf16_t* wc2 = (bf16_t*)alloc((size_t)128 * 1024 * 2);
  bf16_t* wc3 = (bf16_t*)alloc((size_t)256 * 2048 * 2);
  bf16_t* wmuB = (bf16_t*)alloc((size_t)32 * 1024 * 2);
  bf16_t* wlvB = (bf16_t*)alloc((size_t)32 * 1024 * 2);
  bf16_t* wfcB = (bf16_t*)alloc((size_t)1024 * 32 * 2);
  bf16_t* wd0 = (bf16_t*)alloc((size_t)128 * 4096 * 2);
  bf16_t* wd1 = (bf16_t*)alloc((size_t)64 * 2048 * 2);
  bf16_t* wd2 = (bf16_t*)alloc((size_t)32 * 1024 * 2);
  bf16_t* wd3 = (bf16_t*)alloc((size_t)1 * 512 * 2);
  float* asmp = (float*)alloc((size_t)51200 * 4);
  float* dyn  = (float*)alloc((size_t)51 * 512 * 4);
  float* A_t  = (float*)alloc((size_t)50 * 32 * 256 * 4);
  float* C_t  = (float*)alloc((size_t)50 * 32 * 512 * 4);
  float* mu_f = (float*)alloc((size_t)50 * 32 * 16 * 4);
  float* Sg_f = (float*)alloc((size_t)50 * 32 * 256 * 4);
  float* mu_p = (float*)alloc((size_t)50 * 32 * 16 * 4);
  float* Sg_p = (float*)alloc((size_t)50 * 32 * 256 * 4);

  // ---- weight / input conversion (conv+FC weights are already [N][K]) ----
  cvt_f32_bf16<<<cdiv(1638400, 256), 256, 0, stream>>>(x, xb, 1638400);
  cvt_f32_bf16<<<cdiv(512, 256), 256, 0, stream>>>(ew[0], wc0, 512);
  cvt_f32_bf16<<<cdiv(32768, 256), 256, 0, stream>>>(ew[1], wc1, 32768);
  cvt_f32_bf16<<<cdiv(131072, 256), 256, 0, stream>>>(ew[2], wc2, 131072);
  cvt_f32_bf16<<<cdiv(524288, 256), 256, 0, stream>>>(ew[3], wc3, 524288);
  cvt_f32_bf16<<<cdiv(32768, 256), 256, 0, stream>>>(wmu, wmuB, 32768);
  cvt_f32_bf16<<<cdiv(32768, 256), 256, 0, stream>>>(wlv, wlvB, 32768);
  cvt_f32_bf16<<<cdiv(32768, 256), 256, 0, stream>>>(wfc, wfcB, 32768);
  prep_flipw<<<cdiv(128 * 256 * 16, 256), 256, 0, stream>>>(dw[0], wd0, 256, 128);
  prep_flipw<<<cdiv(64 * 128 * 16, 256), 256, 0, stream>>>(dw[1], wd1, 128, 64);
  prep_flipw<<<cdiv(32 * 64 * 16, 256), 256, 0, stream>>>(dw[2], wd2, 64, 32);
  prep_flipw<<<cdiv(1 * 32 * 16, 256), 256, 0, stream>>>(dw[3], wd3, 32, 1);

  // ---- encoder convs (implicit GEMM, bf16 WMMA) ----
  gemm_wmma<1, 1, 0><<<dim3(1, 6400), 128, 0, stream>>>(xb, wc0, eb[0], e1, nullptr, 409600, 32, 16, 1, 32, 32, 16, 16);
  gemm_wmma<1, 1, 0><<<dim3(1, 1600), 128, 0, stream>>>(e1, wc1, eb[1], e2, nullptr, 102400, 64, 512, 32, 16, 16, 8, 8);
  gemm_wmma<1, 1, 0><<<dim3(2, 400), 128, 0, stream>>>(e2, wc2, eb[2], e3, nullptr, 25600, 128, 1024, 64, 8, 8, 4, 4);
  gemm_wmma<1, 1, 0><<<dim3(4, 100), 128, 0, stream>>>(e3, wc3, eb[3], e4, nullptr, 6400, 256, 2048, 128, 4, 4, 2, 2);

  // ---- encoder FC heads ----
  gemm_wmma<0, 0, 1><<<dim3(1, 25), 128, 0, stream>>>(e4, wmuB, bmu, nullptr, amu, 1600, 32, 1024, 1, 1, 1, 1, 1);
  gemm_wmma<0, 0, 1><<<dim3(1, 25), 128, 0, stream>>>(e4, wlvB, blv, nullptr, alv, 1600, 32, 1024, 1, 1, 1, 1, 1);

  // ---- reparameterized sample ----
  sample_kernel<<<cdiv(51200, 256), 256, 0, stream>>>(amu, alv, eps, asmp, abf, 51200);

  // ---- LSTM + mixture + Kalman filter + smoother ----
  lstm_kernel<<<1, 256, 0, stream>>>(asmp, Wih, Whh, bih, bhh, start, dyn);
  mix_kernel<<<1600, 256, 0, stream>>>(dyn, Amat, Cmat, A_t, C_t);
  kalman_kernel<<<32, 256, 0, stream>>>(asmp, A_t, C_t, Qm, Rm, mu1, Sg1, mu_f, Sg_f, mu_p, Sg_p);
  smooth_kernel<<<32, 256, 0, stream>>>(mu_f, Sg_f, mu_p, Sg_p, A_t, musm, sgsm);

  // ---- decoder: FC + deconvs ----
  gemm_wmma<0, 0, 0><<<dim3(16, 25), 128, 0, stream>>>(abf, wfcB, bfc, d0, nullptr, 1600, 1024, 32, 1, 1, 1, 1, 1);
  gemm_wmma<2, 1, 0><<<dim3(2, 400), 128, 0, stream>>>(d0, wd0, db[0], d1, nullptr, 25600, 128, 4096, 256, 2, 2, 4, 4);
  gemm_wmma<2, 1, 0><<<dim3(1, 1600), 128, 0, stream>>>(d1, wd1, db[1], d2, nullptr, 102400, 64, 2048, 128, 4, 4, 8, 8);
  gemm_wmma<2, 1, 0><<<dim3(1, 6400), 128, 0, stream>>>(d2, wd2, db[2], d3, nullptr, 409600, 32, 1024, 64, 8, 8, 16, 16);
  gemm_wmma<2, 2, 1><<<dim3(1, 25600), 128, 0, stream>>>(d3, wd3, db[3], nullptr, xhat, 1638400, 1, 512, 32, 16, 16, 32, 32);
}
